// AuxiliaryEncoder_62328565399963
// MI455X (gfx1250) — matrix operand order repaired and checked by reading, not compiled, not run on hardware
//
#include <hip/hip_runtime.h>

// ---------------------------------------------------------------------------
// Fused 3-layer GAT+FFN encoder for MI455X (gfx1250, wave32, WMMA bf16).
// v3: per K-step, all 6 B fragments are loaded into distinct registers before
//     the 6 WMMAs issue -> clause-grouped loads with progressive
//     s_wait_loadcnt, instead of load->wait0->wmma serialization.
// ---------------------------------------------------------------------------

typedef __attribute__((ext_vector_type(16))) __bf16 v16bf;
typedef __attribute__((ext_vector_type(8)))  __bf16 v8bf;
typedef __attribute__((ext_vector_type(8)))  float  v8f;

#define HH      768
#define H2      1536
#define RR      32          // rows (B*N) per workgroup = 8 samples
#define NTHR    512         // 16 waves
#define XPITCH  776         // fp32 x pitch (768 + 8)
#define BPITCH  784         // bf16 x/h pitch (768 + 16)
#define UPITCH  1552        // bf16 u pitch (1536 + 16)
#define LAYERS  3

// workspace layout (bf16 elements): transposed weights, [N][K] row-major
#define WT_SZ   (768*768)
#define W1T_SZ  (1536*768)
#define W2T_SZ  (768*1536)
#define WT_OFF  0
#define W1T_OFF (3*WT_SZ)
#define W2T_OFF (W1T_OFF + 3*W1T_SZ)

// ---------------------------------------------------------------------------
// Prep: transpose fp32 [K][N] -> bf16 [N][K] (coalesced both sides via LDS)
// ---------------------------------------------------------------------------
__global__ __launch_bounds__(256) void tconv_kernel(const float* __restrict__ src,
                                                    __bf16* __restrict__ dst,
                                                    int K, int N) {
  __shared__ float tile[32][33];
  const int tx = threadIdx.x & 31, ty = threadIdx.x >> 5;  // 32x8
  const int n0 = blockIdx.x * 32, k0 = blockIdx.y * 32;
#pragma unroll
  for (int r = 0; r < 32; r += 8)
    tile[ty + r][tx] = src[(size_t)(k0 + ty + r) * N + (n0 + tx)];
  __syncthreads();
#pragma unroll
  for (int r = 0; r < 32; r += 8)
    dst[(size_t)(n0 + ty + r) * K + (k0 + tx)] = (__bf16)tile[tx][ty + r];
}

// ---------------------------------------------------------------------------
// Fragment builders (contiguous k order, same for A and B; not the ISA table order)
// A 16x32 bf16 from bf16 LDS row: lane L -> M = L%16, half = L/16.
//   elems 0..7  = K[k0 + 16*half + 0..7], elems 8..15 = K[k0+16*half+8+0..7]
// -> two 16-byte ds_load_b128, no conversions.
// ---------------------------------------------------------------------------
__device__ __forceinline__ v16bf make_a_frag(const __bf16* __restrict__ row,
                                             int k0, int half) {
  const __bf16* p = row + k0 + 16 * half;
  v8bf lo = *(const v8bf*)p;
  v8bf hi = *(const v8bf*)(p + 8);
  return __builtin_shufflevector(lo, hi, 0, 1, 2, 3, 4, 5, 6, 7,
                                         8, 9, 10, 11, 12, 13, 14, 15);
}

// B 32x16 bf16 from transposed weights [N][K]: lane L -> N = n0 + L%16,
// elems e = K[k0 + 16*(L/16) + e]  -> 32 contiguous bytes per lane.
__device__ __forceinline__ v16bf load_b_frag(const __bf16* __restrict__ base,
                                             int pitch, int n0, int k0, int lane) {
  const __bf16* p = base + (size_t)(n0 + (lane & 15)) * pitch + k0 + 16 * (lane >> 4);
  __builtin_prefetch(p + 32, 0, 1);   // global_prefetch_b8: next K-block
  return *(const v16bf*)p;
}

__device__ __forceinline__ v8f zero8() {
  v8f z;
#pragma unroll
  for (int e = 0; e < 8; ++e) z[e] = 0.f;
  return z;
}

// One K-step of a 6-column-tile GEMM: load 6 B frags (distinct regs, one
// clause), then 6 WMMAs consuming them with progressive load waits.
__device__ __forceinline__ void gemm6_step(const __bf16* __restrict__ arow,
                                           const __bf16* __restrict__ Bb,
                                           int pitch, int tile0, int k0,
                                           int lane, int half, v8f acc[6]) {
  v16bf af = make_a_frag(arow, k0, half);
  v16bf bf[6];
#pragma unroll
  for (int j = 0; j < 6; ++j)
    bf[j] = load_b_frag(Bb, pitch, (tile0 + j) << 4, k0, lane);
#pragma unroll
  for (int j = 0; j < 6; ++j)
    acc[j] = __builtin_amdgcn_wmma_f32_16x16x32_bf16(
        false, af, false, bf[j], (short)0, acc[j], false, false);
}

// ---------------------------------------------------------------------------
// Fused kernel: one WG = 8 samples (32 rows), 512 threads = 16 waves.
// ---------------------------------------------------------------------------
__global__ __launch_bounds__(512) void fused_encoder_kernel(
    const float* __restrict__ xin,      // [B,4,768]
    const float* __restrict__ lte,      // [4,768]
    const float* __restrict__ att_src,  // [L,4,192]
    const float* __restrict__ att_dst,  // [L,4,192]
    const float* __restrict__ gat_bias, // [L,768]
    const float* __restrict__ ln_g,     // [L,768]
    const float* __restrict__ ln_b,     // [L,768]
    const float* __restrict__ b1,       // [L,1536]
    const float* __restrict__ b2,       // [L,768]
    const __bf16* __restrict__ wsb,     // transposed bf16 weights
    float* __restrict__ out)            // [B,4,768]
{
  __shared__ float  xf [RR][XPITCH];    // x, fp32 (residual / LN precision)
  __shared__ __bf16 xbf[RR][BPITCH];    // x, bf16  (A operand, GEMM 1 & 2)
  __shared__ __bf16 hbf[RR][BPITCH];    // h, bf16  (GEMM1 out, attention src)
  __shared__ __bf16 ubf[RR][UPITCH];    // u, bf16  (A operand, GEMM 3)
  __shared__ float  att_l[2 * HH];      // [src|dst][head*192+d]
  __shared__ float  e_l[8][2][4][4];    // [s][src/dst][j][head]
  __shared__ float  a_l[8][4][4][4];    // [s][i][head][j]

  const int t    = threadIdx.x;
  const int lane = t & 31;
  const int wv   = t >> 5;              // 0..15
  const int half = lane >> 4;
  const int rt   = wv >> 3;             // row tile (0/1) for this wave
  const int cg   = wv & 7;              // column group for this wave
  const int baserow = blockIdx.x * RR;

  // ---- prologue: x = label_embeddings + lte (fp32 + bf16 staging)
  for (int idx = t; idx < RR * HH; idx += NTHR) {
    int r = idx / HH, c = idx - r * HH;
    float v = xin[(size_t)(baserow + r) * HH + c] + lte[(r & 3) * HH + c];
    xf[r][c]  = v;
    xbf[r][c] = (__bf16)v;
  }
  __syncthreads();

  for (int l = 0; l < LAYERS; ++l) {
    // attention vectors for this layer -> LDS (first read is after a barrier)
    for (int idx = t; idx < HH; idx += NTHR) {
      att_l[idx]      = att_src[l * HH + idx];
      att_l[HH + idx] = att_dst[l * HH + idx];
    }

    // ================ stage 1: h = x @ W[l]  -> hbf (bf16) ==================
    {
      const __bf16* Bb = wsb + WT_OFF + (size_t)l * WT_SZ;
      const __bf16* arow = &xbf[rt * 16 + (lane & 15)][0];
      v8f acc[6];
#pragma unroll
      for (int j = 0; j < 6; ++j) acc[j] = zero8();
      for (int k0 = 0; k0 < HH; k0 += 32)
        gemm6_step(arow, Bb, HH, cg * 6, k0, lane, half, acc);
#pragma unroll
      for (int j = 0; j < 6; ++j) {
        int col = ((cg * 6 + j) << 4) + (lane & 15);
        int rb  = rt * 16 + half * 8;            // C: row = e + 8*half
#pragma unroll
        for (int e = 0; e < 8; ++e) hbf[rb + e][col] = (__bf16)acc[j][e];
      }
    }
    __syncthreads();

    // ========== stage 2a: e_src/e_dst dots (192-d, 2 threads/value) ==========
    {
      int part  = t & 1;
      int head  = (t >> 1) & 3;
      int j     = (t >> 3) & 3;
      int which = (t >> 5) & 1;   // 0=src, 1=dst
      int s     = (t >> 6) & 7;
      const __bf16* hrow = &hbf[s * 4 + j][head * 192 + part * 96];
      const float*  av   = &att_l[which * HH + head * 192 + part * 96];
      float p = 0.f;
#pragma unroll 4
      for (int d = 0; d < 96; ++d) p += (float)hrow[d] * av[d];
      p += __shfl_xor(p, 1, 32);
      if (part == 0) e_l[s][which][j][head] = p;
    }
    __syncthreads();

    // ========== stage 2a2: leaky-relu logits + softmax over j (4) ============
    {
      int j = t & 3, i = (t >> 2) & 3, head = (t >> 4) & 3, s = (t >> 6) & 7;
      float lg = e_l[s][1][i][head] + e_l[s][0][j][head];
      lg = lg > 0.f ? lg : 0.2f * lg;
      float m = fmaxf(lg, __shfl_xor(lg, 1, 32));
      m = fmaxf(m, __shfl_xor(m, 2, 32));
      float ex = __expf(lg - m);
      float den = ex;
      den += __shfl_xor(den, 1, 32);
      den += __shfl_xor(den, 2, 32);
      a_l[s][i][head][j] = ex / den;
    }
    __syncthreads();

    // ===== stage 2b: gat aggregate + bias + residual + LayerNorm -> x ========
    {
      int r = t >> 4, lane16 = t & 15;
      int s = r >> 2, i = r & 3;
      const float* gb  = gat_bias + l * HH;
      const float* lg_ = ln_g + l * HH;
      const float* lb_ = ln_b + l * HH;
      float vals[48];
      float sum = 0.f, sq = 0.f;
#pragma unroll
      for (int k = 0; k < 48; ++k) {
        int d = k * 16 + lane16;
        int head = d / 192;
        const float* aw = a_l[s][i][head];
        float g4 = aw[0] * (float)hbf[s * 4 + 0][d]
                 + aw[1] * (float)hbf[s * 4 + 1][d]
                 + aw[2] * (float)hbf[s * 4 + 2][d]
                 + aw[3] * (float)hbf[s * 4 + 3][d];
        float v = g4 + gb[d] + xf[r][d];
        vals[k] = v; sum += v; sq += v * v;
      }
#pragma unroll
      for (int o = 1; o < 16; o <<= 1) {
        sum += __shfl_xor(sum, o, 32);
        sq  += __shfl_xor(sq,  o, 32);
      }
      float mu  = sum * (1.f / HH);
      float var = sq * (1.f / HH) - mu * mu;
      float rs  = rsqrtf(var + 1e-5f);
#pragma unroll
      for (int k = 0; k < 48; ++k) {
        int d = k * 16 + lane16;
        float v = (vals[k] - mu) * rs * lg_[d] + lb_[d];
        xf[r][d]  = v;
        xbf[r][d] = (__bf16)v;
      }
    }
    __syncthreads();

    // ============ stage 3: u = relu(x @ w1 + b1) -> ubf (bf16) ==============
    {
      const __bf16* Bb  = wsb + W1T_OFF + (size_t)l * W1T_SZ;
      const float* b1g  = b1 + l * H2;
      const __bf16* arow = &xbf[rt * 16 + (lane & 15)][0];
      for (int p = 0; p < 2; ++p) {             // 12 col-tiles/wave, 2 passes
        v8f acc[6];
#pragma unroll
        for (int j = 0; j < 6; ++j) acc[j] = zero8();
        for (int k0 = 0; k0 < HH; k0 += 32)
          gemm6_step(arow, Bb, HH, cg * 12 + p * 6, k0, lane, half, acc);
#pragma unroll
        for (int j = 0; j < 6; ++j) {
          int col = ((cg * 12 + p * 6 + j) << 4) + (lane & 15);
          int rb  = rt * 16 + half * 8;
#pragma unroll
          for (int e = 0; e < 8; ++e)
            ubf[rb + e][col] = (__bf16)fmaxf(acc[j][e] + b1g[col], 0.f);
        }
      }
    }
    __syncthreads();

    // ====== stage 4: y = u @ w2 + b2 + x  (pre-LN vals in-place in xf) =======
    {
      const __bf16* Bb  = wsb + W2T_OFF + (size_t)l * W2T_SZ;
      const float* b2g  = b2 + l * HH;
      const __bf16* arow = &ubf[rt * 16 + (lane & 15)][0];
      v8f acc[6];
#pragma unroll
      for (int j = 0; j < 6; ++j) acc[j] = zero8();
      for (int k0 = 0; k0 < H2; k0 += 32)
        gemm6_step(arow, Bb, H2, cg * 6, k0, lane, half, acc);
#pragma unroll
      for (int j = 0; j < 6; ++j) {
        int col = ((cg * 6 + j) << 4) + (lane & 15);
        int rb  = rt * 16 + half * 8;
#pragma unroll
        for (int e = 0; e < 8; ++e) {
          int row = rb + e;                     // (row,col) owned by this lane
          xf[row][col] = acc[j][e] + b2g[col] + xf[row][col];
        }
      }
    }
    __syncthreads();

    // ================== second LayerNorm (same g/b), in-place ================
    {
      int r = t >> 4, lane16 = t & 15;
      const float* lg_ = ln_g + l * HH;
      const float* lb_ = ln_b + l * HH;
      float vals[48];
      float sum = 0.f, sq = 0.f;
#pragma unroll
      for (int k = 0; k < 48; ++k) {
        int d = k * 16 + lane16;
        float v = xf[r][d];
        vals[k] = v; sum += v; sq += v * v;
      }
#pragma unroll
      for (int o = 1; o < 16; o <<= 1) {
        sum += __shfl_xor(sum, o, 32);
        sq  += __shfl_xor(sq,  o, 32);
      }
      float mu  = sum * (1.f / HH);
      float var = sq * (1.f / HH) - mu * mu;
      float rs  = rsqrtf(var + 1e-5f);
#pragma unroll
      for (int k = 0; k < 48; ++k) {
        int d = k * 16 + lane16;
        float v = (vals[k] - mu) * rs * lg_[d] + lb_[d];
        xf[r][d]  = v;
        xbf[r][d] = (__bf16)v;
      }
    }
    __syncthreads();
  }

  // ---- epilogue: write final x (fp32, coalesced)
  for (int idx = t; idx < RR * HH; idx += NTHR) {
    int r = idx / HH, c = idx - r * HH;
    out[(size_t)(baserow + r) * HH + c] = xf[r][c];
  }
}

// ---------------------------------------------------------------------------
extern "C" void kernel_launch(void* const* d_in, const int* in_sizes, int n_in,
                              void* d_out, int out_size, void* d_ws, size_t ws_size,
                              hipStream_t stream) {
  (void)in_sizes; (void)n_in; (void)out_size; (void)ws_size;
  const float* xin      = (const float*)d_in[0];
  const float* lte      = (const float*)d_in[1];
  const float* W        = (const float*)d_in[2];
  const float* att_src  = (const float*)d_in[3];
  const float* att_dst  = (const float*)d_in[4];
  const float* gat_bias = (const float*)d_in[5];
  const float* ln_g     = (const float*)d_in[6];
  const float* ln_b     = (const float*)d_in[7];
  const float* w1       = (const float*)d_in[8];
  const float* b1       = (const float*)d_in[9];
  const float* w2       = (const float*)d_in[10];
  const float* b2       = (const float*)d_in[11];
  float*  out = (float*)d_out;
  __bf16* wsb = (__bf16*)d_ws;

  dim3 blk(256);
  for (int l = 0; l < LAYERS; ++l) {
    tconv_kernel<<<dim3(768 / 32, 768 / 32), blk, 0, stream>>>(
        W + (size_t)l * 768 * 768, wsb + WT_OFF + (size_t)l * WT_SZ, 768, 768);
    tconv_kernel<<<dim3(1536 / 32, 768 / 32), blk, 0, stream>>>(
        w1 + (size_t)l * 768 * 1536, wsb + W1T_OFF + (size_t)l * W1T_SZ, 768, 1536);
    tconv_kernel<<<dim3(768 / 32, 1536 / 32), blk, 0, stream>>>(
        w2 + (size_t)l * 1536 * 768, wsb + W2T_OFF + (size_t)l * W2T_SZ, 1536, 768);
  }

  const int nblocks = (16384 * 4) / RR;   // 2048 workgroups
  fused_encoder_kernel<<<nblocks, dim3(NTHR), 0, stream>>>(
      xin, lte, att_src, att_dst, gat_bias, ln_g, ln_b, b1, b2, wsb, out);
}
